// GATSBlock_2405181685852
// MI455X (gfx1250) — compile-verified, matrix operand order checked by reading, not run on hardware
//
#include <hip/hip_runtime.h>
#include <hip/hip_bf16.h>

// ---------------------------------------------------------------------------
// Types for CDNA5 WMMA (wave32, 16x16x32 bf16 -> f32)
// ---------------------------------------------------------------------------
typedef __attribute__((ext_vector_type(16))) __bf16 v16bf;
typedef __attribute__((ext_vector_type(8)))  __bf16 bf16x8;
typedef __attribute__((ext_vector_type(8)))  float  v8f;

#define BB     2
#define NN_    2048
#define DIMN   1024
#define HEADS  8
#define DHEAD  64
#define INNER  512
#define FF     4096
#define WS     512
#define NWIN   4          // N / WS
#define PADROW 512

// Native f32 -> bf16 conversion (lowers to hardware cvt, not bit-twiddling).
__device__ inline __bf16 f2bf(float f) { return (__bf16)f; }

__device__ inline v8f wmma_bf16(v16bf a, v16bf b, v8f c) {
  return __builtin_amdgcn_wmma_f32_16x16x32_bf16(false, a, false, b, (short)0, c,
                                                 false, false);
}

// Load a v16bf A/B fragment from an LDS row of 32 bf16 (layout: lane half picks
// K in {half*8..half*8+7, 16+half*8..16+half*8+7}).
__device__ inline v16bf lds_frag(const __bf16* rowp, int half) {
  bf16x8 lo = *(const bf16x8*)(rowp + half * 8);
  bf16x8 hi = *(const bf16x8*)(rowp + 16 + half * 8);
  return __builtin_shufflevector(lo, hi, 0,1,2,3,4,5,6,7,8,9,10,11,12,13,14,15);
}

__device__ inline float half_reduce_max(float v) {
  #pragma unroll
  for (int m = 8; m >= 1; m >>= 1) v = fmaxf(v, __shfl_xor(v, m, 32));
  return v;
}
__device__ inline float half_reduce_sum(float v) {
  #pragma unroll
  for (int m = 8; m >= 1; m >>= 1) v += __shfl_xor(v, m, 32);
  return v;
}

// ---------------------------------------------------------------------------
// Generic WMMA GEMM: C = scale*(A @ B) [+ bias] [+ Res]
// A: [M,K] row-major (lda).  B: NN -> [K,N] (ldb=N);  NT -> [N,K] (ldb=K).
// 256 threads = 8 waves; block tile 64x64, BK=32; each wave does 16x32 of C.
// Batched via blockIdx.z decomposed as (zb, zw) with zw < WPB.
// ---------------------------------------------------------------------------
template<bool TRANSB, bool BIAS, bool RES>
__global__ __launch_bounds__(256)
void gemm_kernel(const float* __restrict__ A, const float* __restrict__ B,
                 const float* __restrict__ bias, const float* __restrict__ Res,
                 float* __restrict__ C,
                 int M, int N, int K, int lda, int ldb, int ldc, float scale,
                 int WPB,
                 long long sAb, long long sAw, long long sBb, long long sBw,
                 long long sCb, long long sCw) {
  __shared__ alignas(16) __bf16 As[64][32];
  __shared__ alignas(16) __bf16 Bs[64][32];   // Bs[n][k]

  const int z  = blockIdx.z;
  const int zb = z / WPB, zw = z % WPB;
  A += zb * sAb + zw * sAw;
  B += zb * sBb + zw * sBw;
  C += zb * sCb + zw * sCw;

  const int m0 = blockIdx.y * 64, n0 = blockIdx.x * 64;
  const int tid  = threadIdx.x;
  const int lane = tid & 31, wave = tid >> 5;
  const int half = lane >> 4, lm = lane & 15;
  const int wm = wave & 3, wn = wave >> 2;

  v8f acc0, acc1;
  #pragma unroll
  for (int i = 0; i < 8; ++i) { acc0[i] = 0.f; acc1[i] = 0.f; }

  for (int k0 = 0; k0 < K; k0 += 32) {
    {
      const int row = tid >> 2, seg = tid & 3;
      const float* ap = A + (size_t)(m0 + row) * lda + k0 + seg * 8;
      #pragma unroll
      for (int i = 0; i < 8; ++i) As[row][seg * 8 + i] = f2bf(ap[i]);
      if (k0 + 32 < K) __builtin_prefetch(ap + 32, 0, 0);
    }
    if (TRANSB) {
      const int nrow = tid >> 2, seg = tid & 3;
      const float* bp = B + (size_t)(n0 + nrow) * ldb + k0 + seg * 8;
      #pragma unroll
      for (int i = 0; i < 8; ++i) Bs[nrow][seg * 8 + i] = f2bf(bp[i]);
      if (k0 + 32 < K) __builtin_prefetch(bp + 32, 0, 0);
    } else {
      const int krow = tid >> 3, seg = tid & 7;
      const float* bp = B + (size_t)(k0 + krow) * ldb + n0 + seg * 8;
      #pragma unroll
      for (int i = 0; i < 8; ++i) Bs[seg * 8 + i][krow] = f2bf(bp[i]);
      if (k0 + 32 < K) __builtin_prefetch(bp + (size_t)32 * ldb, 0, 0);
    }
    __syncthreads();

    v16bf a  = lds_frag(&As[wm * 16 + lm][0], half);
    v16bf b0 = lds_frag(&Bs[wn * 32 + lm][0], half);
    v16bf b1 = lds_frag(&Bs[wn * 32 + 16 + lm][0], half);
    acc0 = wmma_bf16(a, b0, acc0);
    acc1 = wmma_bf16(a, b1, acc1);
    __syncthreads();
  }

  #pragma unroll
  for (int r = 0; r < 8; ++r) {
    const int row  = m0 + wm * 16 + r + 8 * half;
    const int col0 = n0 + wn * 32 + lm;
    const int col1 = col0 + 16;
    float v0 = acc0[r] * scale, v1 = acc1[r] * scale;
    if (BIAS) { v0 += bias[col0]; v1 += bias[col1]; }
    if (RES)  { v0 += Res[(size_t)row * ldc + col0];
                v1 += Res[(size_t)row * ldc + col1]; }
    C[(size_t)row * ldc + col0] = v0;
    C[(size_t)row * ldc + col1] = v1;
  }
}

// ---------------------------------------------------------------------------
// Row LayerNorm (optionally pre-applied tanh-GELU). One block per row.
// L in {1024, 4096}; 256 threads; per-thread register staging.
// ---------------------------------------------------------------------------
template<bool GELU>
__global__ __launch_bounds__(256)
void ln_kernel(const float* __restrict__ x, const float* __restrict__ g,
               const float* __restrict__ bt, float* __restrict__ y,
               int L, float eps) {
  __shared__ float red[256];
  const int row = blockIdx.x, tid = threadIdx.x;
  const float* xr = x + (size_t)row * L;
  float* yr = y + (size_t)row * L;
  const int nper = L >> 8;
  float v[16];
  float s = 0.f;
  for (int i = 0; i < nper; ++i) {
    float t = xr[tid + i * 256];
    if (GELU) {
      float t3 = t * t * t;
      t = 0.5f * t * (1.f + tanhf(0.7978845608f * (t + 0.044715f * t3)));
    }
    v[i] = t; s += t;
  }
  red[tid] = s; __syncthreads();
  for (int st = 128; st > 0; st >>= 1) { if (tid < st) red[tid] += red[tid + st]; __syncthreads(); }
  const float mu = red[0] / (float)L;
  __syncthreads();
  float s2 = 0.f;
  for (int i = 0; i < nper; ++i) { float d = v[i] - mu; s2 += d * d; }
  red[tid] = s2; __syncthreads();
  for (int st = 128; st > 0; st >>= 1) { if (tid < st) red[tid] += red[tid + st]; __syncthreads(); }
  const float rstd = rsqrtf(red[0] / (float)L + eps);
  for (int i = 0; i < nper; ++i) {
    const int c = tid + i * 256;
    yr[c] = (v[i] - mu) * rstd * g[c] + bt[c];
  }
}

// RMSNorm over head vectors of length 64; one wave per vector, in place.
__global__ __launch_bounds__(256)
void rmsnorm_kernel(float* __restrict__ x, const float* __restrict__ g,
                    int rowStride) {
  const int lane = threadIdx.x & 31;
  const int wv = blockIdx.x * 8 + (threadIdx.x >> 5);  // vector id
  const int row = wv >> 3, h = wv & 7;
  float* p = x + (size_t)row * rowStride + h * DHEAD;
  float e0 = p[lane], e1 = p[lane + 32];
  float ss = e0 * e0 + e1 * e1;
  #pragma unroll
  for (int m = 16; m >= 1; m >>= 1) ss += __shfl_xor(ss, m, 32);
  const float r = rsqrtf(ss * (1.f / 64.f) + 1e-6f);
  p[lane]      = e0 * r * g[lane];
  p[lane + 32] = e1 * r * g[lane + 32];
}

// Rotary embedding over d=1024 (inv_freq len 512), optional output row pad.
__global__ __launch_bounds__(256)
void rope_kernel(const float* __restrict__ x, float* __restrict__ out,
                 int padRows, float scale) {
  const int r = blockIdx.x;            // 0 .. B*N-1
  const int b = r >> 11, n = r & 2047;
  const float* xi = x + (size_t)r * DIMN;
  float* o = out + ((size_t)b * (NN_ + padRows) + padRows + n) * DIMN;
  for (int j = threadIdx.x; j < DIMN; j += 256) {
    const int f = j & 511;
    const float inv = __expf((float)f * (-2.0f / 1024.0f) * 9.2103403719761836f);
    const float ang = (float)n * inv;
    float sn, cs; __sincosf(ang, &sn, &cs);
    const float rot = (j < 512) ? -xi[j + 512] : xi[j - 512];
    o[j] = (xi[j] * cs + rot * sn) * scale;
  }
}

__global__ __launch_bounds__(256)
void zero_pad_kernel(float* __restrict__ p) {
  const size_t i = (size_t)blockIdx.x * 256 + threadIdx.x;  // < B*512*1024
  const size_t b = i >> 19;
  const size_t rem = i & ((size_t)(PADROW * DIMN) - 1);
  p[b * (size_t)((NN_ + PADROW) * DIMN) + rem] = 0.f;
}

__global__ __launch_bounds__(256)
void vpad_copy_kernel(const float* __restrict__ src, float* __restrict__ dst) {
  const size_t i = (size_t)blockIdx.x * 256 + threadIdx.x;  // < B*N*1024
  const size_t b = i >> 21;
  const size_t rem = i & ((size_t)(NN_ * DIMN) - 1);
  dst[b * (size_t)((NN_ + PADROW) * DIMN) + (size_t)(PADROW * DIMN) + rem] = src[i];
}

// Local-attention mask + softmax over rows of [B*NWIN*WS, 2*WS], in place.
__global__ __launch_bounds__(256)
void softmax_mask_kernel(float* __restrict__ sim) {
  __shared__ float red[256];
  const int rid = blockIdx.x, tid = threadIdx.x;
  const int rem = rid & 2047;
  const int w = rem >> 9, i = rem & 511;
  const int qpos = w * WS + i;
  float* row = sim + (size_t)rid * 1024;
  float v[4];
  float mx = -3.0e38f;
  #pragma unroll
  for (int t = 0; t < 4; ++t) {
    const int j = tid + t * 256;
    const int kpos = w * WS - WS + j;
    float s = row[j];
    const bool masked = (kpos < 0) || (qpos < kpos);
    s = masked ? -1e9f : s;
    v[t] = s; mx = fmaxf(mx, s);
  }
  red[tid] = mx; __syncthreads();
  for (int st = 128; st > 0; st >>= 1) { if (tid < st) red[tid] = fmaxf(red[tid], red[tid + st]); __syncthreads(); }
  const float rmax = red[0];
  __syncthreads();
  float sum = 0.f;
  #pragma unroll
  for (int t = 0; t < 4; ++t) { v[t] = __expf(v[t] - rmax); sum += v[t]; }
  red[tid] = sum; __syncthreads();
  for (int st = 128; st > 0; st >>= 1) { if (tid < st) red[tid] += red[tid + st]; __syncthreads(); }
  const float inv = 1.f / red[0];
  #pragma unroll
  for (int t = 0; t < 4; ++t) row[tid + t * 256] = v[t] * inv;
}

// ---------------------------------------------------------------------------
// Flash-style causal attention, qk already rms-normed.
// q: [B*N, 512] (h*64+d), kv: [B*N, 1024] (k cols 0..511, v cols 512..1023).
// 256 threads = 8 waves; each wave owns 16 query rows; 32 keys per step.
// Uniform per-block key loop so __syncthreads() is legal around P staging.
// ---------------------------------------------------------------------------
__global__ __launch_bounds__(256)
void flash_attn_kernel(const float* __restrict__ q, const float* __restrict__ kv,
                       float* __restrict__ o) {
  __shared__ alignas(16) __bf16 Pb[8][16][32];
  const int lane = threadIdx.x & 31, wave = threadIdx.x >> 5;
  const int half = lane >> 4, lm = lane & 15;
  const int b = blockIdx.z, h = blockIdx.y;
  const int q0blk = blockIdx.x * 128;
  const int q0 = q0blk + wave * 16;

  // Q fragments (K-dim = 64 -> 2 chunks), held in registers for the whole loop
  v16bf qa0, qa1;
  {
    const float* qrow = q + ((size_t)(b * NN_ + q0 + lm) * INNER) + h * DHEAD;
    #pragma unroll
    for (int e = 0; e < 16; ++e) {
      const int grp = e >> 3, wq = e & 7;
      const int d0 = grp * 16 + half * 8 + wq;
      qa0[e] = f2bf(qrow[d0]);
      qa1[e] = f2bf(qrow[32 + d0]);
    }
  }

  v8f oacc[4];
  #pragma unroll
  for (int t = 0; t < 4; ++t)
    #pragma unroll
    for (int r = 0; r < 8; ++r) oacc[t][r] = 0.f;
  float mrow[8], lrow[8], alpha[8];
  #pragma unroll
  for (int r = 0; r < 8; ++r) { mrow[r] = -3.0e38f; lrow[r] = 0.f; }

  const int kmax = q0blk + 127;
  for (int kb = 0; kb <= kmax; kb += 32) {
    // S = Q @ K^T for two 16-key tiles
    v8f s0, s1;
    #pragma unroll
    for (int r = 0; r < 8; ++r) { s0[r] = 0.f; s1[r] = 0.f; }
    #pragma unroll
    for (int c = 0; c < 2; ++c) {
      v16bf kf0, kf1;
      const float* kr0 = kv + ((size_t)(b * NN_ + kb + lm) * 1024) + h * DHEAD + c * 32;
      const float* kr1 = kr0 + (size_t)16 * 1024;
      #pragma unroll
      for (int e = 0; e < 16; ++e) {
        const int grp = e >> 3, wq = e & 7;
        const int d = grp * 16 + half * 8 + wq;
        kf0[e] = f2bf(kr0[d]);
        kf1[e] = f2bf(kr1[d]);
      }
      const v16bf aq = (c == 0) ? qa0 : qa1;
      s0 = wmma_bf16(aq, kf0, s0);
      s1 = wmma_bf16(aq, kf1, s1);
    }
    // online softmax (rows live in one 16-lane half)
    #pragma unroll
    for (int r = 0; r < 8; ++r) {
      const int aq  = q0 + r + 8 * half;
      const int ak0 = kb + lm, ak1 = ak0 + 16;
      const float v0 = (ak0 <= aq) ? s0[r] * 0.125f : -1e9f;
      const float v1 = (ak1 <= aq) ? s1[r] * 0.125f : -1e9f;
      const float mx = half_reduce_max(fmaxf(v0, v1));
      const float nm = fmaxf(mrow[r], mx);
      alpha[r] = __expf(mrow[r] - nm);
      mrow[r] = nm;
      const float e0 = __expf(v0 - nm), e1 = __expf(v1 - nm);
      lrow[r] = lrow[r] * alpha[r] + half_reduce_sum(e0 + e1);
      const int m = r + 8 * half;
      Pb[wave][m][lm]      = f2bf(e0);
      Pb[wave][m][16 + lm] = f2bf(e1);
    }
    __syncthreads();
    // P (C-layout) -> A-layout fragment via LDS
    v16bf pa;
    #pragma unroll
    for (int e = 0; e < 16; ++e) {
      const int grp = e >> 3, wq = e & 7;
      const int kk = grp * 16 + half * 8 + wq;
      pa[e] = Pb[wave][lm][kk];
    }
    #pragma unroll
    for (int t = 0; t < 4; ++t)
      #pragma unroll
      for (int r = 0; r < 8; ++r) oacc[t][r] *= alpha[r];
    // O += P @ V (K=32 keys, N=64 head dim -> 4 tiles)
    #pragma unroll
    for (int t = 0; t < 4; ++t) {
      v16bf vb;
      #pragma unroll
      for (int e = 0; e < 16; ++e) {
        const int grp = e >> 3, wq = e & 7;
        const int kk = grp * 16 + half * 8 + wq;
        const float* vr = kv + ((size_t)(b * NN_ + kb + kk) * 1024) + 512 + h * DHEAD;
        vb[e] = f2bf(vr[t * 16 + lm]);
      }
      oacc[t] = wmma_bf16(pa, vb, oacc[t]);
    }
    __syncthreads();
  }
  #pragma unroll
  for (int t = 0; t < 4; ++t)
    #pragma unroll
    for (int r = 0; r < 8; ++r) {
      const int row = b * NN_ + q0 + r + 8 * half;
      const int col = h * DHEAD + t * 16 + lm;
      o[(size_t)row * INNER + col] = oacc[t][r] / lrow[r];
    }
}

// ---------------------------------------------------------------------------
// Host-side orchestration
// ---------------------------------------------------------------------------
extern "C" void kernel_launch(void* const* d_in, const int* in_sizes, int n_in,
                              void* d_out, int out_size, void* d_ws, size_t ws_size,
                              hipStream_t stream) {
  (void)in_sizes; (void)n_in; (void)out_size; (void)ws_size;
  const float* text  = (const float*)d_in[0];
  const float* audio = (const float*)d_in[1];
  const float* video = (const float*)d_in[2];
  const float* Wa    = (const float*)d_in[3];
  const float* ba    = (const float*)d_in[4];
  const float* lna_g = (const float*)d_in[5];
  const float* lna_b = (const float*)d_in[6];
  const float* Wvid  = (const float*)d_in[7];
  const float* bvid  = (const float*)d_in[8];
  const float* lnv_g = (const float*)d_in[9];
  const float* lnv_b = (const float*)d_in[10];
  const float* ln1_g = (const float*)d_in[11];
  const float* ln1_b = (const float*)d_in[12];
  const float* Wq    = (const float*)d_in[13];
  const float* Wkv   = (const float*)d_in[14];
  const float* qn_g  = (const float*)d_in[15];
  const float* kn_g  = (const float*)d_in[16];
  const float* Wo    = (const float*)d_in[17];
  const float* W1    = (const float*)d_in[18];
  const float* b1    = (const float*)d_in[19];
  const float* ln2_g = (const float*)d_in[20];
  const float* ln2_b = (const float*)d_in[21];
  const float* W2    = (const float*)d_in[22];
  const float* b2    = (const float*)d_in[23];
  float* out = (float*)d_out;
  float* ws  = (float*)d_ws;

  const int M = BB * NN_;  // 4096 token rows
  // Workspace layout (floats) with buffer reuse:
  float* A0 = ws;                       // 4194304 : audio_t -> sim -> o
  float* A1 = ws + 4194304;             // 4194304 : video_t -> xloc -> x2
  float* A2 = ws + 8388608;             // 4194304 : q_rope -> h_ln1
  float* A3 = ws + 12582912;            // 5242880 : k_pad  -> q
  float* A4 = ws + 17825792;            // 5242880 : v_pad  -> kv
  float* A5 = ws + 23068672;            // 16777216: ffn hidden (in-place gelu+ln)

  const long long Z = 0;
  const long long sQb = (long long)NN_ * DIMN;         // 2097152
  const long long sQw = (long long)WS * DIMN;          // 524288
  const long long sPb = (long long)(NN_ + PADROW) * DIMN; // 2621440
  const long long sSb = 4LL * sQw;                     // sim batch stride

  // 1) adapters: layernorm(audio@Wa+ba), layernorm(video@Wvid+bvid)
  gemm_kernel<false, true, false><<<dim3(16, 64, 1), 256, 0, stream>>>(
      audio, Wa, ba, nullptr, A0, M, DIMN, DIMN, DIMN, DIMN, DIMN, 1.f,
      1, Z, Z, Z, Z, Z, Z);
  ln_kernel<false><<<M, 256, 0, stream>>>(A0, lna_g, lna_b, A0, DIMN, 1e-5f);
  gemm_kernel<false, true, false><<<dim3(16, 64, 1), 256, 0, stream>>>(
      video, Wvid, bvid, nullptr, A1, M, DIMN, DIMN, DIMN, DIMN, DIMN, 1.f,
      1, Z, Z, Z, Z, Z, Z);
  ln_kernel<false><<<M, 256, 0, stream>>>(A1, lnv_g, lnv_b, A1, DIMN, 1e-5f);

  // 2) rotary + padded K/V windows
  rope_kernel<<<M, 256, 0, stream>>>(text, A2, 0, 0.03125f);   // q * d^-0.5
  zero_pad_kernel<<<(BB * PADROW * DIMN) / 256, 256, 0, stream>>>(A3);
  zero_pad_kernel<<<(BB * PADROW * DIMN) / 256, 256, 0, stream>>>(A4);
  rope_kernel<<<M, 256, 0, stream>>>(A0, A3, PADROW, 1.0f);    // k_pad
  vpad_copy_kernel<<<(M * DIMN) / 256, 256, 0, stream>>>(A1, A4);

  // 3) local attention: S = Q @ Kwin^T (batched over B*NWIN), softmax, P@Vwin
  gemm_kernel<true, false, false><<<dim3(16, 8, BB * NWIN), 256, 0, stream>>>(
      A2, A3, nullptr, nullptr, A0, WS, 2 * WS * 1, DIMN, DIMN, DIMN, 1024, 1.f,
      NWIN, sQb, sQw, sPb, sQw, sSb, sQw);
  softmax_mask_kernel<<<BB * NWIN * WS, 256, 0, stream>>>(A0);
  gemm_kernel<false, false, false><<<dim3(16, 8, BB * NWIN), 256, 0, stream>>>(
      A0, A4, nullptr, nullptr, A1, WS, DIMN, 2 * WS, 1024, DIMN, DIMN, 1.f,
      NWIN, sSb, sQw, sPb, sQw, sQb, sQw);

  // 4) ln1, q/kv projections, qk rmsnorm
  ln_kernel<false><<<M, 256, 0, stream>>>(A1, ln1_g, ln1_b, A2, DIMN, 1e-5f);
  gemm_kernel<false, false, false><<<dim3(8, 64, 1), 256, 0, stream>>>(
      A2, Wq, nullptr, nullptr, A3, M, INNER, DIMN, DIMN, INNER, INNER, 1.f,
      1, Z, Z, Z, Z, Z, Z);
  gemm_kernel<false, false, false><<<dim3(16, 64, 1), 256, 0, stream>>>(
      A2, Wkv, nullptr, nullptr, A4, M, 2 * INNER, DIMN, DIMN, 2 * INNER,
      2 * INNER, 1.f, 1, Z, Z, Z, Z, Z, Z);
  rmsnorm_kernel<<<(M * HEADS) / 8, 256, 0, stream>>>(A3, qn_g, INNER);
  rmsnorm_kernel<<<(M * HEADS) / 8, 256, 0, stream>>>(A4, kn_g, 2 * INNER);

  // 5) causal flash attention -> o
  flash_attn_kernel<<<dim3(NN_ / 128, HEADS, BB), 256, 0, stream>>>(A3, A4, A0);

  // 6) x = 2 * (o @ Wo)   (fuses the reference's x = x + x)
  gemm_kernel<false, false, false><<<dim3(16, 64, 1), 256, 0, stream>>>(
      A0, Wo, nullptr, nullptr, A1, M, DIMN, INNER, INNER, DIMN, DIMN, 2.f,
      1, Z, Z, Z, Z, Z, Z);

  // 7) FFN: layernorm(gelu(x@W1+b1)) @ W2 + b2 + x
  gemm_kernel<false, true, false><<<dim3(64, 64, 1), 256, 0, stream>>>(
      A1, W1, b1, nullptr, A5, M, FF, DIMN, DIMN, FF, FF, 1.f,
      1, Z, Z, Z, Z, Z, Z);
  ln_kernel<true><<<M, 256, 0, stream>>>(A5, ln2_g, ln2_b, A5, FF, 1e-5f);
  gemm_kernel<false, true, true><<<dim3(16, 64, 1), 256, 0, stream>>>(
      A5, W2, b2, A1, out, M, DIMN, FF, FF, DIMN, DIMN, 1.f,
      1, Z, Z, Z, Z, Z, Z);
}
